// QuantumDataReuploadModel_3959959847454
// MI455X (gfx1250) — compile-verified, matrix operand order checked by reading, not run on hardware
//
#include <hip/hip_runtime.h>

// ---------------------------------------------------------------------------
// QuantumDataReuploadModel for MI455X (gfx1250, wave32)
//   Stage 1: per-batch 12-qubit statevector sim, state resident in LDS.
//            Gates on disjoint wires commute -> wires 0-4 (flat bits 11-7)
//            are register-local under idx = j*128 + t and are fused into the
//            CNOT-permutation register pass. Only wires 5-11 touch LDS pairs.
//   Stage 2: fused 2-layer MLP via v_wmma_f32_16x16x32_f16 (fp32 accum).
// ---------------------------------------------------------------------------

#define NQ        12
#define DIM       4096                 // 2^12 amplitudes
#define NLAYERS   3
#define BLK1      128                  // sim threads per block (4 waves)
#define APT       (DIM / BLK1)         // 32 amplitudes / thread
#define PAIRS     (DIM / 2)            // 2048 pairs per 1q gate
#define PPT       (PAIRS / BLK1)       // 16 pairs / thread
#define BATCH     2048
#define BLK2      256                  // MLP threads per block (8 waves)

typedef __attribute__((ext_vector_type(16))) _Float16 v16h;
typedef __attribute__((ext_vector_type(8)))  float    v8f;

// K index for the 16-bit 16x32 A / 32x16 B WMMA fragment layout (ISA 7.12.2):
// VGPR v holds halves (h=0 low, h=1 high); lanes 0-15 (g=0) vs 16-31 (g=1).
__device__ __forceinline__ int kmap(int v, int h, int g) {
    int base = (v < 4) ? (2 * v) : (16 + 2 * (v - 4));
    return base + h + 8 * g;
}

// complex 2x2 gate on an amplitude pair
__device__ __forceinline__ void gate2(float2& a0, float2& a1,
                                      float2 u00, float2 u01,
                                      float2 u10, float2 u11) {
    float2 n0, n1;
    n0.x = u00.x * a0.x - u00.y * a0.y + u01.x * a1.x - u01.y * a1.y;
    n0.y = u00.x * a0.y + u00.y * a0.x + u01.x * a1.y + u01.y * a1.x;
    n1.x = u10.x * a0.x - u10.y * a0.y + u11.x * a1.x - u11.y * a1.y;
    n1.y = u10.x * a0.y + u10.y * a0.x + u11.x * a1.y + u11.y * a1.x;
    a0 = n0;
    a1 = n1;
}

// ---------------------------------------------------------------------------
// Kernel 1: quantum circuit simulation, one block per batch element
// ---------------------------------------------------------------------------
__global__ __launch_bounds__(BLK1) void qsim_kernel(
    const float* __restrict__ x,    // [B,48]
    const float* __restrict__ W,    // [12,3,4]
    const float* __restrict__ Bp,   // [12,3]
    float* __restrict__ qz)         // [B,12]
{
    __shared__ float2 sS[DIM];        // statevector (32 KB)
    __shared__ float2 sU[NQ][4];      // U3 matrices (layer-invariant)
    __shared__ float  sP[NQ * 3];     // params
    __shared__ float  sZ[BLK1][NQ];   // reduction scratch

    const int t = threadIdx.x;
    const int b = blockIdx.x;

    // params[c,p] = dot4(W[c,p,:], x[b, c*4 : c*4+4]) + bias[c,p]
    if (t < NQ * 3) {
        int c = t / 3, p = t % 3;
        const float* w  = W + (c * 3 + p) * 4;
        const float* xc = x + b * 48 + c * 4;
        sP[t] = w[0] * xc[0] + w[1] * xc[1] + w[2] * xc[2] + w[3] * xc[3] + Bp[t];
    }
    __syncthreads();

    // U3 = [[ct, -e^{il} st], [e^{ip} st, e^{i(p+l)} ct]]  (same every layer)
    if (t < NQ) {
        float th = sP[t * 3 + 0], ph = sP[t * 3 + 1], lm = sP[t * 3 + 2];
        float ct, st, cp, sp, cl, sl;
        __sincosf(0.5f * th, &st, &ct);
        __sincosf(ph, &sp, &cp);
        __sincosf(lm, &sl, &cl);
        sU[t][0] = make_float2(ct, 0.f);
        sU[t][1] = make_float2(-cl * st, -sl * st);
        sU[t][2] = make_float2(cp * st,  sp * st);
        float cpl = cp * cl - sp * sl;
        float spl = sp * cl + cp * sl;
        sU[t][3] = make_float2(cpl * ct, spl * ct);
    }

    // |0...0>
    for (int j = 0; j < APT; ++j) {
        int i = j * BLK1 + t;
        sS[i] = make_float2(i == 0 ? 1.f : 0.f, 0.f);
    }
    __syncthreads();

    for (int layer = 0; layer < NLAYERS; ++layer) {
        // ---- wires 5..11 (flat bits 6..0): cross-lane, via LDS pairs ----
        for (int w = 5; w < NQ; ++w) {
            const int pos   = 11 - w;             // 6..0
            const int lmask = (1 << pos) - 1;
            const float2 u00 = sU[w][0], u01 = sU[w][1];
            const float2 u10 = sU[w][2], u11 = sU[w][3];
            for (int j = 0; j < PPT; ++j) {
                int p  = j * BLK1 + t;
                int i0 = ((p & ~lmask) << 1) | (p & lmask);
                int i1 = i0 | (1 << pos);
                float2 a0 = sS[i0], a1 = sS[i1];
                gate2(a0, a1, u00, u01, u10, u11);
                sS[i0] = a0;
                sS[i1] = a1;
            }
            __syncthreads();
        }

        // ---- register pass: wires 0..4 (flat bits 11..7 == reg index j),
        //      then the composed CNOT-ring permutation scatter ----
        float2 regs[APT];
        #pragma unroll
        for (int j = 0; j < APT; ++j) regs[j] = sS[j * BLK1 + t];

        #pragma unroll
        for (int w = 0; w < 5; ++w) {             // bit (4-w) of j
            const int rb    = 4 - w;
            const int rmask = (1 << rb) - 1;
            const float2 u00 = sU[w][0], u01 = sU[w][1];
            const float2 u10 = sU[w][2], u11 = sU[w][3];
            #pragma unroll
            for (int jj = 0; jj < APT / 2; ++jj) {
                int j0 = ((jj & ~rmask) << 1) | (jj & rmask);
                int j1 = j0 | (1 << rb);
                gate2(regs[j0], regs[j1], u00, u01, u10, u11);
            }
        }

        __syncthreads();   // every thread finished reading its sources
        #pragma unroll
        for (int j = 0; j < APT; ++j) {
            unsigned v = (unsigned)(j * BLK1 + t);
            #pragma unroll
            for (int i = 0; i < NQ; ++i) {        // CNOT(i, i+1 mod 12) chain
                int pc = 11 - i;
                int pt = 11 - ((i + 1) % NQ);
                v ^= ((v >> pc) & 1u) << pt;      // if control set, flip target
            }
            sS[v] = regs[j];
        }
        __syncthreads();
    }

    // ---- PauliZ expectations ----
    float z[NQ];
    #pragma unroll
    for (int i = 0; i < NQ; ++i) z[i] = 0.f;
    for (int j = 0; j < APT; ++j) {
        int idx = j * BLK1 + t;
        float2 a = sS[idx];
        float p = a.x * a.x + a.y * a.y;
        #pragma unroll
        for (int i = 0; i < NQ; ++i)
            z[i] += ((idx >> (11 - i)) & 1) ? -p : p;
    }
    #pragma unroll
    for (int i = 0; i < NQ; ++i) sZ[t][i] = z[i];
    __syncthreads();
    if (t < NQ) {
        float acc = 0.f;
        for (int j = 0; j < BLK1; ++j) acc += sZ[j][t];
        qz[b * NQ + t] = acc;
    }
}

// ---------------------------------------------------------------------------
// Kernel 2: fused MLP  out = relu(qz @ w1^T + b1) @ w2^T + b2   via WMMA
// Block: 16 batch rows, 8 waves. Waves 0-3 build h[16x64] (K=12 padded to 32),
// all 8 waves produce out[16x256] (K=64 = 2 WMMA steps), 2 col-tiles per wave.
// ---------------------------------------------------------------------------
__global__ __launch_bounds__(BLK2) void mlp_kernel(
    const float* __restrict__ qz,   // [B,12]
    const float* __restrict__ w1,   // [64,12]
    const float* __restrict__ b1,   // [64]
    const float* __restrict__ w2,   // [256,64]
    const float* __restrict__ b2,   // [256]
    float* __restrict__ out)        // [B,256]
{
    __shared__ float    sQ[16][12];
    __shared__ _Float16 sH[16][68];   // padded rows to dodge bank conflicts

    const int t    = threadIdx.x;
    const int wv   = t >> 5;
    const int lane = t & 31;
    const int g    = lane >> 4;       // half-wave group
    const int n16  = lane & 15;
    const int r0   = blockIdx.x * 16;

    if (t < 192) {
        int m = t / 12, k = t % 12;
        sQ[m][k] = qz[(r0 + m) * 12 + k];
    }
    __syncthreads();

    // ---- GEMM1: h = relu(qz @ w1^T + b1), waves 0..3, 16 cols each ----
    if (wv < 4) {
        v16h a, bm;
        #pragma unroll
        for (int j = 0; j < 16; ++j) {
            int k = kmap(j >> 1, j & 1, g);
            a[j] = (k < 12) ? (_Float16)sQ[n16][k] : (_Float16)0.f;
        }
        int n = wv * 16 + n16;
        #pragma unroll
        for (int j = 0; j < 16; ++j) {
            int k = kmap(j >> 1, j & 1, g);
            bm[j] = (k < 12) ? (_Float16)w1[n * 12 + k] : (_Float16)0.f;
        }
        float bn = b1[n];
        v8f c = {bn, bn, bn, bn, bn, bn, bn, bn};
        c = __builtin_amdgcn_wmma_f32_16x16x32_f16(false, a, false, bm,
                                                   (short)0, c, false, false);
        #pragma unroll
        for (int r = 0; r < 8; ++r) {
            float h = c[r] > 0.f ? c[r] : 0.f;     // relu
            sH[r + 8 * g][n] = (_Float16)h;
        }
    }
    __syncthreads();

    // ---- GEMM2: out = h @ w2^T + b2, all 8 waves, K = 64 ----
    v16h a0, a1;
    #pragma unroll
    for (int j = 0; j < 16; ++j) {
        int k = kmap(j >> 1, j & 1, g);
        a0[j] = sH[n16][k];
        a1[j] = sH[n16][32 + k];
    }
    #pragma unroll
    for (int tt = 0; tt < 2; ++tt) {
        int n = (2 * wv + tt) * 16 + n16;
        const float* w2row = w2 + n * 64;
        v16h bm0, bm1;
        #pragma unroll
        for (int j = 0; j < 16; ++j) {
            int k = kmap(j >> 1, j & 1, g);
            bm0[j] = (_Float16)w2row[k];
            bm1[j] = (_Float16)w2row[32 + k];
        }
        float bn = b2[n];
        v8f c = {bn, bn, bn, bn, bn, bn, bn, bn};
        c = __builtin_amdgcn_wmma_f32_16x16x32_f16(false, a0, false, bm0,
                                                   (short)0, c, false, false);
        c = __builtin_amdgcn_wmma_f32_16x16x32_f16(false, a1, false, bm1,
                                                   (short)0, c, false, false);
        #pragma unroll
        for (int r = 0; r < 8; ++r)
            out[(r0 + r + 8 * g) * 256 + n] = c[r];
    }
}

// ---------------------------------------------------------------------------
extern "C" void kernel_launch(void* const* d_in, const int* in_sizes, int n_in,
                              void* d_out, int out_size, void* d_ws, size_t ws_size,
                              hipStream_t stream) {
    const float* x  = (const float*)d_in[0];   // [2048,48]
    const float* W  = (const float*)d_in[1];   // [12,3,4]
    const float* Bp = (const float*)d_in[2];   // [12,3]
    const float* w1 = (const float*)d_in[3];   // [64,12]
    const float* b1 = (const float*)d_in[4];   // [64]
    const float* w2 = (const float*)d_in[5];   // [256,64]
    const float* b2 = (const float*)d_in[6];   // [256]
    float* outp = (float*)d_out;               // [2048,256]
    float* qzws = (float*)d_ws;                // [2048,12] scratch

    qsim_kernel<<<BATCH, BLK1, 0, stream>>>(x, W, Bp, qzws);
    mlp_kernel<<<BATCH / 16, BLK2, 0, stream>>>(qzws, w1, b1, w2, b2, outp);
}
